// SDEABlock_50714973831712
// MI455X (gfx1250) — matrix operand order repaired and checked by reading, not compiled, not run on hardware
//
#include <hip/hip_runtime.h>
#include <hip/hip_bf16.h>

typedef _Float16 f16_t;
typedef __attribute__((ext_vector_type(16))) _Float16 v16h;
typedef __attribute__((ext_vector_type(8)))  _Float16 v8h;
typedef __attribute__((ext_vector_type(8)))  float    v8f;

#define BB 4
#define CC 32
#define HH 256
#define WW 512
#define TW 64
#define LW 66      // TW + 2 halo
#define MAXD 48

// ---------------------------------------------------------------------------
// Prep: swizzle conv weights (OIHW f32) into WMMA A-fragment layout (f16) and
// fold bias+BN into per-channel alpha/beta.
// A-frag layout per conv: [mt(2)][tap(9)][lane(32)][e(16)] halfs, where
//   lane<16 : M = mt*16+lane, K(e) = e      ({0..15})
//   lane>=16: M = mt*16+lane-16, K(e) = e+16 ({16..31})
// (contiguous k = 16*lanehalf + e; B reads ci in the same order). K index == ci for tap-GEMM.
// ---------------------------------------------------------------------------
__global__ void prep_kernel(const float* __restrict__ w1, const float* __restrict__ w2,
    const float* c1b, const float* g1g, const float* g1b, const float* g1m, const float* g1v,
    const float* c2b, const float* g2g, const float* g2bb, const float* g2m, const float* g2v,
    f16_t* __restrict__ wfrag, float* __restrict__ alpha, float* __restrict__ beta)
{
  int tid = blockIdx.x * 256 + threadIdx.x;
  if (tid < 2 * 2 * 9 * 32 * 16) {
    int c    = tid / 9216;         // which conv
    int r    = tid % 9216;
    int e    = r & 15;
    int lane = (r >> 4) & 31;
    int t    = (r >> 9) % 9;
    int mt   = r / 4608;
    int lhalf = lane >> 4, lm = lane & 15;
    int k  = e + lhalf * 16;   // ci
    int co = mt * 16 + lm;
    int kh = t / 3, kw = t % 3;
    const float* src = (c == 0) ? w1 : w2;
    wfrag[tid] = (f16_t)src[((co * CC + k) * 3 + kh) * 3 + kw];
  }
  if (tid < 64) {
    int c = tid >> 5, ch = tid & 31;
    float g  = c ? g2g[ch]  : g1g[ch];
    float vv = c ? g2v[ch]  : g1v[ch];
    float mm = c ? g2m[ch]  : g1m[ch];
    float bb = c ? g2bb[ch] : g1b[ch];
    float bi = c ? c2b[ch]  : c1b[ch];
    float s = g * rsqrtf(vv + 1e-5f);
    alpha[tid] = s;
    beta[tid]  = (bi - mm) * s + bb;
  }
}

// ---------------------------------------------------------------------------
// Fused 3x3 conv + bias + BN + ReLU via WMMA implicit GEMM.
// Block = 128 threads (4 waves). Tile: 32 co x 64 w of one (b,h) row.
// Wave (mt = wid>>1, nt = wid&1) owns 16 co x 32 w = two 16x16 accumulators.
// K loop = 9 taps x one v_wmma_f32_16x16x32_f16 k-step (K=32 = C_IN) each.
// Input staged in LDS as [3 rows][66 w][32 ci] f16 (HWC) -> B-fragment is a
// contiguous 32B read per lane. Output stored NHWC f16 (16B store per acc).
// ---------------------------------------------------------------------------
template<bool IN_F16>
__global__ void __launch_bounds__(128)
conv3x3_bn_relu(const float* __restrict__ inF32,   // NCHW f32 (if !IN_F16)
                const f16_t* __restrict__ inF16,   // NHWC f16 (if IN_F16)
                const f16_t* __restrict__ wfrag,   // A-fragments for this conv
                const float* __restrict__ alpha,
                const float* __restrict__ beta,
                f16_t* __restrict__ out)           // NHWC f16
{
  __shared__ __align__(64) f16_t lds[3 * LW * CC];
  const int tid = threadIdx.x;
  const int b = blockIdx.z, h = blockIdx.y, w0 = blockIdx.x * TW;

  if (IN_F16) {
    for (int p = tid; p < 3 * LW; p += 128) {
      int hr = p / LW, wl = p % LW;
      int gh = h + hr - 1, gw = w0 + wl - 1;
      v16h lo = {}; v16h hi = {};
      if (gh >= 0 && gh < HH && gw >= 0 && gw < WW) {
        const f16_t* src = inF16 + (((size_t)(b * HH + gh)) * WW + gw) * CC;
        lo = *(const v16h*)src;
        hi = *(const v16h*)(src + 16);
      }
      *(v16h*)&lds[(hr * LW + wl) * CC]      = lo;
      *(v16h*)&lds[(hr * LW + wl) * CC + 16] = hi;
    }
  } else {
    for (int i = tid; i < CC * 3 * LW; i += 128) {
      int ci = i / (3 * LW);
      int r  = i % (3 * LW);
      int hr = r / LW, wl = r % LW;
      int gh = h + hr - 1, gw = w0 + wl - 1;
      float v = 0.f;
      if (gh >= 0 && gh < HH && gw >= 0 && gw < WW)
        v = inF32[(((size_t)(b * CC + ci)) * HH + gh) * WW + gw];
      lds[(hr * LW + wl) * CC + ci] = (f16_t)v;
    }
  }
  __syncthreads();

  const int wid = tid >> 5, lane = tid & 31;
  const int mt = wid >> 1, nt = wid & 1;
  const int lhalf = lane >> 4, lm = lane & 15;

  v8f acc0 = {0.f, 0.f, 0.f, 0.f, 0.f, 0.f, 0.f, 0.f};
  v8f acc1 = acc0;

  const v16h* af = (const v16h*)wfrag + (size_t)mt * 9 * 32 + lane;
  #pragma unroll
  for (int t = 0; t < 9; ++t) {
    const int kh = t / 3, kw = t - kh * 3;
    v16h a = af[t * 32];
    // B 32x16 layout: lanes 0-15 K=0..15 (ci 0..15), lanes 16-31 K=16..31
    const int base = nt * 32 + lm + kw;   // lds w index of input pixel
    v16h b0 = *(const v16h*)&lds[((kh * LW) + base) * CC + lhalf * 16];
    v16h b1 = *(const v16h*)&lds[((kh * LW) + base + 16) * CC + lhalf * 16];
    acc0 = __builtin_amdgcn_wmma_f32_16x16x32_f16(false, a, false, b0, (short)0, acc0, false, false);
    acc1 = __builtin_amdgcn_wmma_f32_16x16x32_f16(false, a, false, b1, (short)0, acc1, false, false);
  }

  // Epilogue: D element r -> co = mt*16 + (lane>>4)*8 + r, pixel = n0 + (lane&15)
  const int coBase = mt * 16 + lhalf * 8;
  float al[8], be[8];
  #pragma unroll
  for (int r2 = 0; r2 < 8; ++r2) { al[r2] = alpha[coBase + r2]; be[r2] = beta[coBase + r2]; }

  #pragma unroll
  for (int s = 0; s < 2; ++s) {
    v8f acc = s ? acc1 : acc0;
    v8h o;
    #pragma unroll
    for (int r2 = 0; r2 < 8; ++r2) {
      float f = acc[r2] * al[r2] + be[r2];
      o[r2] = (f16_t)fmaxf(f, 0.f);
    }
    const int wpix = w0 + nt * 32 + s * 16 + lm;
    *(v8h*)&out[(((size_t)(b * HH + h)) * WW + wpix) * CC + coBase] = o;
  }
}

// ---------------------------------------------------------------------------
// g2: 1x1 conv 32->1 over NHWC f16 (contiguous 32-ch reduction per pixel).
// ---------------------------------------------------------------------------
__global__ void g2_kernel(const f16_t* __restrict__ g1L, const f16_t* __restrict__ g1R,
                          const float* __restrict__ g2w, const float* __restrict__ g2b,
                          float* __restrict__ outL, float* __restrict__ outR)
{
  const int NP = BB * HH * WW;
  int idx = blockIdx.x * 256 + threadIdx.x;
  if (idx >= 2 * NP) return;
  int side = idx / NP;
  int p = idx - side * NP;
  const f16_t* g = (side ? g1R : g1L) + (size_t)p * CC;
  float s = g2b[0];
  #pragma unroll
  for (int ci = 0; ci < CC; ++ci) s += (float)g[ci] * g2w[ci];
  (side ? outR : outL)[p] = s;
}

// ---------------------------------------------------------------------------
// Disparity cost (min |r[w-48+j] - l[w]| over j=0..95, inf-padded), sigmoid
// gate, residual combine. Output NCHW f32 (left then right).
// ---------------------------------------------------------------------------
__global__ void __launch_bounds__(128)
final_kernel(const float* __restrict__ g2L, const float* __restrict__ g2R,
             const f16_t* __restrict__ g1L, const f16_t* __restrict__ g1R,
             const float* __restrict__ left, const float* __restrict__ right,
             float* __restrict__ out)
{
  __shared__ float rrow[128 + 2 * MAXD];
  const int tid = threadIdx.x;
  const int w0 = blockIdx.x * 128;
  const int h = blockIdx.y;
  const int b = blockIdx.z >> 1;
  const int side = blockIdx.z & 1;
  const float* gl = side ? g2R : g2L;   // "l" stream of weight_matrix
  const float* gr = side ? g2L : g2R;   // "r" stream (inf-padded)
  const int rowBase = (b * HH + h) * WW;

  for (int i = tid; i < 128 + 2 * MAXD; i += 128) {
    int gw = w0 - MAXD + i;
    rrow[i] = (gw >= 0 && gw < WW) ? gr[rowBase + gw] : __builtin_inff();
  }
  __syncthreads();

  const int w = w0 + tid;
  const float l = gl[rowBase + w];
  float cost = __builtin_inff();
  #pragma unroll 4
  for (int j = 0; j < 2 * MAXD; ++j)
    cost = fminf(cost, fabsf(rrow[tid + j] - l));
  const float wt = 1.f / (1.f + __expf(-cost));

  const f16_t* g1 = (side ? g1R : g1L) + ((size_t)rowBase + w) * CC;
  const float* xin = side ? right : left;
  float* o = out + (size_t)side * ((size_t)BB * CC * HH * WW);
  #pragma unroll
  for (int co = 0; co < CC; ++co) {
    size_t off = (((size_t)(b * CC + co)) * HH + h) * WW + w;
    o[off] = (float)g1[co] * wt + xin[off];
  }
}

// ---------------------------------------------------------------------------
extern "C" void kernel_launch(void* const* d_in, const int* in_sizes, int n_in,
                              void* d_out, int out_size, void* d_ws, size_t ws_size,
                              hipStream_t stream)
{
  const float* left  = (const float*)d_in[0];
  const float* right = (const float*)d_in[1];
  const float* c1w = (const float*)d_in[2];
  const float* c1b = (const float*)d_in[3];
  const float* b1g = (const float*)d_in[4];
  const float* b1b = (const float*)d_in[5];
  const float* b1m = (const float*)d_in[6];
  const float* b1v = (const float*)d_in[7];
  const float* c2w = (const float*)d_in[8];
  const float* c2b = (const float*)d_in[9];
  const float* b2g = (const float*)d_in[10];
  const float* b2b = (const float*)d_in[11];
  const float* b2m = (const float*)d_in[12];
  const float* b2v = (const float*)d_in[13];
  const float* g2w = (const float*)d_in[14];
  const float* g2b = (const float*)d_in[15];

  char* ws = (char*)d_ws;
  const size_t NPIX = (size_t)BB * HH * WW;           // 524288 pixels
  const size_t G1B  = NPIX * CC * sizeof(f16_t);      // 32 MiB per image (f16 NHWC)
  f16_t* wfrag = (f16_t*)(ws);                        // 36864 B (both convs)
  float* alpha = (float*)(ws + 36864);                // 256 B
  float* beta  = (float*)(ws + 37120);                // 256 B
  f16_t* g1a   = (f16_t*)(ws + 37376);                // conv1 output scratch (reused L/R)
  f16_t* g1L   = (f16_t*)(ws + 37376 + G1B);
  f16_t* g1R   = (f16_t*)(ws + 37376 + 2 * G1B);
  float* g2L   = (float*)(ws + 37376 + 3 * G1B);
  float* g2R   = (float*)(ws + 37376 + 3 * G1B + NPIX * sizeof(float));

  prep_kernel<<<72, 256, 0, stream>>>(c1w, c2w, c1b, b1g, b1b, b1m, b1v,
                                      c2b, b2g, b2b, b2m, b2v, wfrag, alpha, beta);

  dim3 cgrid(WW / TW, HH, BB);
  conv3x3_bn_relu<false><<<cgrid, 128, 0, stream>>>(left,  nullptr, wfrag,        alpha,      beta,      g1a);
  conv3x3_bn_relu<true ><<<cgrid, 128, 0, stream>>>(nullptr, g1a,   wfrag + 9216, alpha + 32, beta + 32, g1L);
  conv3x3_bn_relu<false><<<cgrid, 128, 0, stream>>>(right, nullptr, wfrag,        alpha,      beta,      g1a);
  conv3x3_bn_relu<true ><<<cgrid, 128, 0, stream>>>(nullptr, g1a,   wfrag + 9216, alpha + 32, beta + 32, g1R);

  g2_kernel<<<(int)((2 * NPIX + 255) / 256), 256, 0, stream>>>(g1L, g1R, g2w, g2b, g2L, g2R);

  dim3 fgrid(WW / 128, HH, BB * 2);
  final_kernel<<<fgrid, 128, 0, stream>>>(g2L, g2R, g1L, g1R, left, right, (float*)d_out);
}
